// ScaledDotProductAttention_27951647162684
// MI455X (gfx1250) — matrix operand order repaired and checked by reading, not compiled, not run on hardware
//
#include <hip/hip_runtime.h>
#include <hip/hip_bf16.h>

typedef __attribute__((ext_vector_type(16))) _Float16 v16h;
typedef __attribute__((ext_vector_type(8)))  _Float16 half8;
typedef __attribute__((ext_vector_type(8)))  float    v8f;

#define WMMA_F16(A, B, C) \
    __builtin_amdgcn_wmma_f32_16x16x32_f16(false, (A), false, (B), (short)0, (C), false, false)

// Problem constants
#define BATCH 4
#define SEQ   2048
#define DIM   1024
#define DK    64
#define ROWS  (BATCH * SEQ)   // 8192

// ---------------------------------------------------------------------------
// Kernel 0: convert & transpose weights -> f16 Wt[3][64][1024]
// Wt[(w*64+n)*1024 + k] = W_w[k*64 + n]
// ---------------------------------------------------------------------------
__global__ __launch_bounds__(256) void prep_weights(const float* __restrict__ Wq,
                                                    const float* __restrict__ Wk,
                                                    const float* __restrict__ Wv,
                                                    _Float16* __restrict__ Wt) {
    int idx = blockIdx.x * 256 + threadIdx.x;
    if (idx >= 3 * DK * DIM) return;
    int w   = idx / (DK * DIM);
    int rem = idx % (DK * DIM);
    int n   = rem / DIM;
    int k   = rem % DIM;
    const float* W = (w == 0) ? Wq : (w == 1) ? Wk : Wv;
    Wt[idx] = (_Float16)W[(size_t)k * DK + n];
}

__device__ inline void cvt8(const float4& a, const float4& b, _Float16* dst) {
    dst[0] = (_Float16)a.x; dst[1] = (_Float16)a.y;
    dst[2] = (_Float16)a.z; dst[3] = (_Float16)a.w;
    dst[4] = (_Float16)b.x; dst[5] = (_Float16)b.y;
    dst[6] = (_Float16)b.z; dst[7] = (_Float16)b.w;
}

// ---------------------------------------------------------------------------
// Kernel 1: QKV projections via WMMA, register double-buffered over K-steps.
// 4 waves/block, each wave = 16 rows x 64 cols of one of {Q,K,V}.
// Q is pre-scaled by log2(e)/sqrt(dk) (base-2 softmax downstream).
//   Qh,Kh: [8192][64] f16 row-major.  Vt: [B][64][T] f16 (transposed).
// ---------------------------------------------------------------------------
__global__ __launch_bounds__(128) void qkv_proj(const float* __restrict__ x,
                                                const _Float16* __restrict__ Wt,
                                                _Float16* __restrict__ Qh,
                                                _Float16* __restrict__ Kh,
                                                _Float16* __restrict__ Vt) {
    const int w    = blockIdx.y;                       // 0=Q 1=K 2=V
    const int warp = threadIdx.x >> 5;
    const int lane = threadIdx.x & 31;
    const int lm   = lane & 15;
    const int hi   = lane >> 4;                        // 0 | 1
    const int m0   = (blockIdx.x * 4 + warp) * 16;     // tile row base
    const int m    = m0 + lm;                          // this lane's A row

    const float*    xrow  = x  + (size_t)m * DIM + hi * 16;
    const _Float16* Wrow0 = Wt + (size_t)w * DK * DIM + (size_t)( 0 + lm) * DIM + hi * 16;
    const _Float16* Wrow1 = Wt + (size_t)w * DK * DIM + (size_t)(16 + lm) * DIM + hi * 16;
    const _Float16* Wrow2 = Wt + (size_t)w * DK * DIM + (size_t)(32 + lm) * DIM + hi * 16;
    const _Float16* Wrow3 = Wt + (size_t)w * DK * DIM + (size_t)(48 + lm) * DIM + hi * 16;

    v8f c0 = {}, c1 = {}, c2 = {}, c3 = {};

    // ---- prologue: load k-step 0 ----
    float4 f0 = *(const float4*)(xrow);
    float4 f1 = *(const float4*)(xrow + 4);
    float4 f2 = *(const float4*)(xrow + 8);
    float4 f3 = *(const float4*)(xrow + 12);
    v16h nb0 = *(const v16h*)(Wrow0);
    v16h nb1 = *(const v16h*)(Wrow1);
    v16h nb2 = *(const v16h*)(Wrow2);
    v16h nb3 = *(const v16h*)(Wrow3);

    for (int k0 = 0; k0 < DIM; k0 += 32) {
        // capture current operands
        float4 g0 = f0, g1 = f1, g2 = f2, g3 = f3;
        v16h   b0 = nb0, b1 = nb1, b2 = nb2, b3 = nb3;

        // prefetch next k-step into the staging registers
        const int kn = k0 + 32;
        if (kn < DIM) {
            f0  = *(const float4*)(xrow + kn);
            f1  = *(const float4*)(xrow + kn + 4);
            f2  = *(const float4*)(xrow + kn + 8);
            f3  = *(const float4*)(xrow + kn + 12);
            nb0 = *(const v16h*)(Wrow0 + kn);
            nb1 = *(const v16h*)(Wrow1 + kn);
            nb2 = *(const v16h*)(Wrow2 + kn);
            nb3 = *(const v16h*)(Wrow3 + kn);
        }

        // convert A to f16 and run 4 WMMAs on current operands
        v16h a;
        _Float16* ap = (_Float16*)&a;
        cvt8(g0, g1, ap);
        cvt8(g2, g3, ap + 8);

        c0 = WMMA_F16(a, b0, c0);
        c1 = WMMA_F16(a, b1, c1);
        c2 = WMMA_F16(a, b2, c2);
        c3 = WMMA_F16(a, b3, c3);
    }

    if (w == 0 || w == 1) {
        // 1/sqrt(64) * log2(e) folded into Q for base-2 flash softmax
        const float qscale = (w == 0) ? 0.125f * 1.44269504088896f : 1.0f;
        _Float16* out = (w == 0) ? Qh : Kh;
#pragma unroll
        for (int r = 0; r < 8; ++r) {
            int mm = m0 + r + hi * 8;
            _Float16* orow = out + (size_t)mm * DK;
            orow[ 0 + lm] = (_Float16)(c0[r] * qscale);
            orow[16 + lm] = (_Float16)(c1[r] * qscale);
            orow[32 + lm] = (_Float16)(c2[r] * qscale);
            orow[48 + lm] = (_Float16)(c3[r] * qscale);
        }
    } else {
        // V transposed: Vt[(b*64 + d)*T + t]; lane holds fixed d, 8 consecutive t
        const int b  = m0 >> 11;                 // /2048
        const int t0 = (m0 & (SEQ - 1)) + hi * 8;
#pragma unroll
        for (int nsub = 0; nsub < 4; ++nsub) {
            const v8f& c = (nsub == 0) ? c0 : (nsub == 1) ? c1 : (nsub == 2) ? c2 : c3;
            int d = nsub * 16 + lm;
            _Float16* dst = Vt + ((size_t)(b * DK + d)) * SEQ + t0;
#pragma unroll
            for (int r = 0; r < 8; ++r) dst[r] = (_Float16)c[r];
        }
    }
}

// ---------------------------------------------------------------------------
// Kernel 2: causal flash attention (base-2 online softmax). 4 waves/block,
// each wave owns 16 query rows; key blocks of 32.
// ---------------------------------------------------------------------------
__global__ __launch_bounds__(128) void attn_fwd(const _Float16* __restrict__ Qh,
                                                const _Float16* __restrict__ Kh,
                                                const _Float16* __restrict__ Vt,
                                                float* __restrict__ out) {
    __shared__ __align__(32) _Float16 Plds[4][16 * 32];   // per-wave P staging

    const int warp = threadIdx.x >> 5;
    const int lane = threadIdx.x & 31;
    const int lm   = lane & 15;
    const int hi   = lane >> 4;

    const int tile = blockIdx.x * 4 + warp;    // 0 .. 511
    const int b    = tile >> 7;                // 128 tiles per batch
    const int m0   = (tile & 127) * 16;        // query row base within batch

    const _Float16* Qrow  = Qh + ((size_t)(b * SEQ + m0 + lm)) * DK;
    const _Float16* Kbase = Kh + (size_t)b * SEQ * DK;
    const _Float16* Vbase = Vt + (size_t)b * DK * SEQ;
    _Float16* Pw = &Plds[warp][0];

    // A operands for Q: kstep0 (features 0..31) and kstep1 (32..63)
    v16h aq0, aq1;
    {
        half8* p0 = (half8*)&aq0;
        half8* p1 = (half8*)&aq1;
        p0[0] = *(const half8*)(Qrow + hi * 16);
        p0[1] = *(const half8*)(Qrow + hi * 16 + 8);
        p1[0] = *(const half8*)(Qrow + 32 + hi * 16);
        p1[1] = *(const half8*)(Qrow + 32 + hi * 16 + 8);
    }

    float mrow[8], lrow[8];
#pragma unroll
    for (int r = 0; r < 8; ++r) { mrow[r] = -1e30f; lrow[r] = 0.0f; }
    v8f o0 = {}, o1 = {}, o2 = {}, o3 = {};

    const int nkb = (m0 + 16 + 31) >> 5;       // causal: keys 0 .. m0+15

    for (int jb = 0; jb < nkb; ++jb) {
        const int n0 = jb * 32;

        const _Float16* kr0 = Kbase + (size_t)(n0 + lm) * DK + hi * 16;
        const _Float16* kr1 = Kbase + (size_t)(n0 + 16 + lm) * DK + hi * 16;
        const _Float16* vr  = Vbase + (size_t)lm * SEQ + n0 + hi * 16;

        // issue V loads early: they are consumed only after the softmax VALU work
        v16h bv0 = *(const v16h*)(vr);
        v16h bv1 = *(const v16h*)(vr + (size_t)16 * SEQ);
        v16h bv2 = *(const v16h*)(vr + (size_t)32 * SEQ);
        v16h bv3 = *(const v16h*)(vr + (size_t)48 * SEQ);

        // prefetch next key block's K/V lines into cache (global_prefetch_b8)
        if (n0 + 32 < m0 + 16) {
            __builtin_prefetch(kr0 + 32 * DK, 0, 3);
            __builtin_prefetch(kr1 + 32 * DK, 0, 3);
            __builtin_prefetch(vr + 32, 0, 3);
        }

        // ---- S = Q @ K^T (two 16-col subtiles, K-dim = 64 in 2 steps) ----
        v16h b00 = *(const v16h*)(kr0);
        v16h b01 = *(const v16h*)(kr0 + 32);
        v16h b10 = *(const v16h*)(kr1);
        v16h b11 = *(const v16h*)(kr1 + 32);

        v8f s0 = {}, s1 = {};
        s0 = WMMA_F16(aq0, b00, s0);
        s0 = WMMA_F16(aq1, b01, s0);
        s1 = WMMA_F16(aq0, b10, s1);
        s1 = WMMA_F16(aq1, b11, s1);

        // ---- mask + base-2 online softmax (rows live in 16-lane halves) ----
#pragma unroll
        for (int r = 0; r < 8; ++r) {
            const int mg = m0 + r + hi * 8;           // global query row (in batch)
            float v0 = s0[r];
            float v1 = s1[r];
            if (n0 + lm      > mg) v0 = -1e30f;       // causal mask
            if (n0 + 16 + lm > mg) v1 = -1e30f;

            float rm = fmaxf(v0, v1);
            rm = fmaxf(rm, __shfl_xor(rm, 1, 32));
            rm = fmaxf(rm, __shfl_xor(rm, 2, 32));
            rm = fmaxf(rm, __shfl_xor(rm, 4, 32));
            rm = fmaxf(rm, __shfl_xor(rm, 8, 32));

            float mn    = fmaxf(mrow[r], rm);
            float alpha = exp2f(mrow[r] - mn);        // S already in log2 domain
            float p0    = exp2f(v0 - mn);
            float p1    = exp2f(v1 - mn);

            float rs = p0 + p1;
            rs += __shfl_xor(rs, 1, 32);
            rs += __shfl_xor(rs, 2, 32);
            rs += __shfl_xor(rs, 4, 32);
            rs += __shfl_xor(rs, 8, 32);

            lrow[r] = lrow[r] * alpha + rs;
            mrow[r] = mn;
            o0[r] *= alpha; o1[r] *= alpha; o2[r] *= alpha; o3[r] *= alpha;

            // stage P tile (C-layout -> row-major LDS)
            const int row = r + hi * 8;
            Pw[row * 32 + lm]      = (_Float16)p0;
            Pw[row * 32 + 16 + lm] = (_Float16)p1;
        }

        // wave-local LDS RAW: wait for DS stores, block scheduler reordering
        __builtin_amdgcn_wave_barrier();
        asm volatile("s_wait_dscnt 0" ::: "memory");

        // reload P as A-operand (16x32, K=key index)
        v16h ap;
        {
            half8* pp = (half8*)&ap;
            pp[0] = *(const half8*)(Pw + lm * 32 + hi * 16);
            pp[1] = *(const half8*)(Pw + lm * 32 + hi * 16 + 8);
        }
        __builtin_amdgcn_wave_barrier();

        // ---- O += P @ V ----
        o0 = WMMA_F16(ap, bv0, o0);
        o1 = WMMA_F16(ap, bv1, o1);
        o2 = WMMA_F16(ap, bv2, o2);
        o3 = WMMA_F16(ap, bv3, o3);

        // ensure A-operand reads finished before next iteration overwrites LDS
        asm volatile("s_wait_dscnt 0" ::: "memory");
    }

    // ---- normalize and store context [B,T,64] f32 ----
#pragma unroll
    for (int r = 0; r < 8; ++r) {
        const float inv = 1.0f / lrow[r];
        const int   mm  = b * SEQ + m0 + r + hi * 8;
        float* dst = out + (size_t)mm * DK;
        dst[ 0 + lm] = o0[r] * inv;
        dst[16 + lm] = o1[r] * inv;
        dst[32 + lm] = o2[r] * inv;
        dst[48 + lm] = o3[r] * inv;
    }
}

// ---------------------------------------------------------------------------
extern "C" void kernel_launch(void* const* d_in, const int* in_sizes, int n_in,
                              void* d_out, int out_size, void* d_ws, size_t ws_size,
                              hipStream_t stream) {
    const float* x  = (const float*)d_in[0];
    const float* Wq = (const float*)d_in[1];
    const float* Wk = (const float*)d_in[2];
    const float* Wv = (const float*)d_in[3];

    char* ws = (char*)d_ws;
    _Float16* Wt = (_Float16*)ws;                        // 3*64*1024*2  = 384 KB
    _Float16* Qh = (_Float16*)(ws + 3 * DK * DIM * 2);   // 8192*64*2    = 1 MB
    _Float16* Kh = Qh + (size_t)ROWS * DK;               // 1 MB
    _Float16* Vt = Kh + (size_t)ROWS * DK;               // 1 MB  (total ~3.4 MB)

    // weights: 3*64*1024 = 196608 elements
    prep_weights<<<(3 * DK * DIM + 255) / 256, 256, 0, stream>>>(Wq, Wk, Wv, Wt);

    // projections: 512 row-tiles x 3 matrices; 4 waves/block
    qkv_proj<<<dim3(ROWS / 16 / 4, 3), 128, 0, stream>>>(x, Wt, Qh, Kh, Vt);

    // attention: 512 query tiles; 4 waves/block
    attn_fwd<<<ROWS / 16 / 4, 128, 0, stream>>>(Qh, Kh, Vt, (float*)d_out);
}
